// GBP_70428873720122
// MI455X (gfx1250) — matrix operand order repaired and checked by reading, not compiled, not run on hardware
//
#include <hip/hip_runtime.h>
#include <hip/hip_bf16.h>
#include <cstdint>
#include <math.h>

// ---------------------------------------------------------------------------
// CDNA5 (gfx1250) fused GVP layer: all GEMMs on v_wmma_f32_16x16x32_bf16.
// - B operands pre-swizzled to WMMA lane layout in d_ws (bf16), streamed from
//   L2 in groups of <=8 tiles; outer loops pinned (#pragma unroll 1) so the
//   compiler cannot unroll-and-jam or hoist whole-K weight sets (no spills).
// - Dual accumulators per output tile (even/odd k-tiles) -> 2 independent
//   WMMA chains per wave to cover XDL latency at 2 waves/SIMD occupancy.
// ---------------------------------------------------------------------------

typedef __attribute__((ext_vector_type(16))) __bf16 v16bf;
typedef __attribute__((ext_vector_type(8)))  float  v8f;

#define WMMA_BF16(a, b, c) \
  __builtin_amdgcn_wmma_f32_16x16x32_bf16(false, (a), false, (b), (short)0, (c), false, false)

#define V8F_ZERO {0.f, 0.f, 0.f, 0.f, 0.f, 0.f, 0.f, 0.f}

// ---- A-operand load from LDS, 16-bit A 16x32 layout (ISA 7.12.2):
// lanes 0-15: M=lane,   K = k0+0..7 (v0-3) and k0+16..23 (v4-7)
// lanes16-31: M=lane-16,K = k0+8..15          and k0+24..31
__device__ __forceinline__ v16bf lds_loadA(const __bf16* base, int rowStride,
                                           int m0, int k0, int lane) {
  int m  = m0 + (lane & 15);
  int hi = lane >> 4;
  const __bf16* p = base + m * rowStride + k0 + hi * 8;
  union { uint4 u[2]; v16bf v; } cv;
  cv.u[0] = *(const uint4*)(p);
  cv.u[1] = *(const uint4*)(p + 16);
  return cv.v;
}

// ---- B-operand load from pre-swizzled weights in d_ws:
// tile = 512 bf16; lane owns 16 contiguous bf16 (32B) -> 2x global b128.
__device__ __forceinline__ v16bf glb_loadB(const __bf16* __restrict__ ws,
                                           int tile, int lane) {
  const __bf16* p = ws + ((long)tile << 9) + (lane << 4);
  union { uint4 u[2]; v16bf v; } cv;
  cv.u[0] = ((const uint4*)p)[0];
  cv.u[1] = ((const uint4*)p)[1];
  return cv.v;
}

// ---------------------------------------------------------------------------
// Weight conversion/swizzle: W [Nout][K] row-major f32 -> bf16 tiles.
// Tile t = nt*Ktiles + kt. Within a tile: lane l, j=0..15 ->
//   n = nt*16 + (l&15), k = kt*32 + (l>>4)*8 + (j&7) + (j>>3)*16   (WMMA B lane layout)
// ---------------------------------------------------------------------------
__global__ void cvt_weights(const float* __restrict__ W, __bf16* __restrict__ dst,
                            int K, int Ktiles, int total) {
  int e = blockIdx.x * 256 + threadIdx.x;
  if (e >= total) return;
  int t    = e >> 9;
  int r    = e & 511;
  int lane = r >> 4;
  int j    = r & 15;
  int nt   = t / Ktiles;
  int kt   = t - nt * Ktiles;
  int n    = nt * 16 + (lane & 15);
  int k    = kt * 32 + ((lane >> 4) << 3) + (j & 7) + ((j >> 3) << 4);
  dst[e] = (__bf16)W[(long)n * K + k];
}

// ---------------------------------------------------------------------------
// Fused kernel. Block = 64 rows of N, 256 threads (8 wave32).
// wave w: M-tile mt = w&3 (rows mt*16..mt*16+15), column half ch = w>>2.
//
// LDS (bytes), rows padded +8 bf16 for bank-conflict-free b128 A loads:
//   [0      ) aTri : bf16 [3][64][264]  = 101376   (later: sS bf16[64][520],
//                                                   then gSig f32[64][264])
//   [101376 ) sVH  : bf16 [3][64][264]  = 101376
//   [202752 ) sVN  : bf16 [64][264]     =  33792
//   [236544 ) sSIG : bf16 [64][520]     =  66560
//   total 303104 B (<= 320 KB WGP LDS)
// ---------------------------------------------------------------------------
__global__ __launch_bounds__(256, 1) void gvp_fused(
    const float* __restrict__ s, const float* __restrict__ v,
    const float* __restrict__ bs, const float* __restrict__ bg,
    const __bf16* __restrict__ wsWd, const __bf16* __restrict__ wsWs,
    const __bf16* __restrict__ wsWu, const __bf16* __restrict__ wsWg,
    float* __restrict__ out_sr, float* __restrict__ out_vr, int N) {
  extern __shared__ char smem[];
  __bf16* aTri = (__bf16*)(smem);            // phase0/1 A operands (v by component)
  __bf16* sVH  = (__bf16*)(smem + 101376);   // vh, kept for phase 5
  __bf16* sVN  = (__bf16*)(smem + 202752);   // vnorm
  __bf16* sSIG = (__bf16*)(smem + 236544);   // sigmoid(sr)
  __bf16* sS   = aTri;                       // overlay: staged s (phase 3)
  float*  gSig = (float*)(smem);             // overlay: sigmoid(gate) (phase 5)

  const int tid  = threadIdx.x;
  const int lane = tid & 31;
  const int wave = tid >> 5;
  const int mt   = wave & 3;
  const int ch   = wave >> 2;
  const int hi   = lane >> 4;
  const int n0   = blockIdx.x * 64;
  const int m0   = mt * 16;
  const int colL = lane & 15;

  // ---- Phase 0: stage v tile -> aTri[c][row][i] (bf16), coalesced reads
  for (int idx = tid; idx < 64 * 768; idx += 256) {
    int row = idx / 768;
    int j   = idx - row * 768;
    int i   = j / 3;
    int c   = j - i * 3;
    float val = (n0 + row < N) ? v[(long)(n0 + row) * 768 + j] : 0.0f;
    aTri[c * 16896 + row * 264 + i] = (__bf16)val;
  }
  __syncthreads();

  // ---- Phase 1: vh_c = v_c @ Wd^T   (K=256). B-tiles of Wd are shared by
  //      all three components: preload 8 tiles once, reuse 3x.
#pragma unroll 1
  for (int nt8 = 0; nt8 < 8; ++nt8) {
    int nt = ch * 8 + nt8;
    v16bf bq[8];
#pragma unroll
    for (int kt = 0; kt < 8; ++kt) bq[kt] = glb_loadB(wsWd, nt * 8 + kt, lane);
    int col = nt * 16 + colL;
#pragma unroll 1
    for (int c = 0; c < 3; ++c) {
      const __bf16* A = aTri + c * 16896;
      v8f ac0 = V8F_ZERO, ac1 = V8F_ZERO;
#pragma unroll
      for (int kt = 0; kt < 8; kt += 2) {
        v16bf a0 = lds_loadA(A, 264, m0, kt * 32, lane);
        v16bf a1 = lds_loadA(A, 264, m0, (kt + 1) * 32, lane);
        ac0 = WMMA_BF16(a0, bq[kt], ac0);
        ac1 = WMMA_BF16(a1, bq[kt + 1], ac1);
      }
      v8f acc = ac0 + ac1;
#pragma unroll
      for (int r = 0; r < 8; ++r)
        sVH[c * 16896 + (m0 + hi * 8 + r) * 264 + col] = (__bf16)acc[r];
    }
  }
  __syncthreads();

  // ---- Phase 2: vnorm = sqrt(sum_c vh^2)+eps -> sVN ; stage s -> sS (bf16)
  for (int idx = tid; idx < 64 * 256; idx += 256) {
    int row = idx >> 8, h = idx & 255;
    float x0 = (float)sVH[0 * 16896 + row * 264 + h];
    float x1 = (float)sVH[1 * 16896 + row * 264 + h];
    float x2 = (float)sVH[2 * 16896 + row * 264 + h];
    sVN[row * 264 + h] = (__bf16)(sqrtf(x0 * x0 + x1 * x1 + x2 * x2) + 1e-8f);
  }
  for (int idx = tid; idx < 64 * 512; idx += 256) {
    int row = idx >> 9, k = idx & 511;
    float val = (n0 + row < N) ? s[(long)(n0 + row) * 512 + k] : 0.0f;
    sS[row * 520 + k] = (__bf16)val;
  }
  __syncthreads();

  // ---- Phase 3: sr = [s | vnorm] @ Ws^T + bs  (K = 512 + 256 = 24 k-tiles)
  //      Weights streamed in groups of 4 tiles to bound VGPR pressure.
#pragma unroll 1
  for (int nt16 = 0; nt16 < 16; ++nt16) {
    int nt = ch * 16 + nt16;
    v8f ac0 = V8F_ZERO, ac1 = V8F_ZERO;
#pragma unroll 1
    for (int g = 0; g < 4; ++g) {          // s part: k-tiles 0..15
      v16bf bq[4];
#pragma unroll
      for (int q = 0; q < 4; ++q) bq[q] = glb_loadB(wsWs, nt * 24 + g * 4 + q, lane);
      __builtin_prefetch((const void*)(wsWs + ((long)(nt * 24 + g * 4 + 4) << 9) + (lane << 4)), 0, 1);
#pragma unroll
      for (int q = 0; q < 4; q += 2) {
        v16bf a0 = lds_loadA(sS, 520, m0, (g * 4 + q) * 32, lane);
        v16bf a1 = lds_loadA(sS, 520, m0, (g * 4 + q + 1) * 32, lane);
        ac0 = WMMA_BF16(a0, bq[q], ac0);
        ac1 = WMMA_BF16(a1, bq[q + 1], ac1);
      }
    }
#pragma unroll 1
    for (int g = 0; g < 2; ++g) {          // vnorm part: k-tiles 16..23
      v16bf bq[4];
#pragma unroll
      for (int q = 0; q < 4; ++q) bq[q] = glb_loadB(wsWs, nt * 24 + 16 + g * 4 + q, lane);
#pragma unroll
      for (int q = 0; q < 4; q += 2) {
        v16bf a0 = lds_loadA(sVN, 264, m0, (g * 4 + q) * 32, lane);
        v16bf a1 = lds_loadA(sVN, 264, m0, (g * 4 + q + 1) * 32, lane);
        ac0 = WMMA_BF16(a0, bq[q], ac0);
        ac1 = WMMA_BF16(a1, bq[q + 1], ac1);
      }
    }
    v8f acc = ac0 + ac1;
    int col = nt * 16 + colL;
    float bias = bs[col];
#pragma unroll
    for (int r = 0; r < 8; ++r) {
      int row = m0 + hi * 8 + r;
      float val = acc[r] + bias;
      float sg  = 1.0f / (1.0f + __expf(-val));
      sSIG[row * 520 + col] = (__bf16)sg;
      if (n0 + row < N) out_sr[(long)(n0 + row) * 512 + col] = fmaxf(val, 0.0f);
    }
  }
  __syncthreads();

  // ---- Phase 4: gate = sigmoid(sr) @ Wg^T + bg ; sigmoid -> gSig (f32, overlays aTri)
#pragma unroll 1
  for (int nt8 = 0; nt8 < 8; ++nt8) {
    int nt = ch * 8 + nt8;
    v8f ac0 = V8F_ZERO, ac1 = V8F_ZERO;
#pragma unroll 1
    for (int g = 0; g < 4; ++g) {          // K = 512 -> 16 k-tiles, 4 groups
      v16bf bq[4];
#pragma unroll
      for (int q = 0; q < 4; ++q) bq[q] = glb_loadB(wsWg, nt * 16 + g * 4 + q, lane);
#pragma unroll
      for (int q = 0; q < 4; q += 2) {
        v16bf a0 = lds_loadA(sSIG, 520, m0, (g * 4 + q) * 32, lane);
        v16bf a1 = lds_loadA(sSIG, 520, m0, (g * 4 + q + 1) * 32, lane);
        ac0 = WMMA_BF16(a0, bq[q], ac0);
        ac1 = WMMA_BF16(a1, bq[q + 1], ac1);
      }
    }
    v8f acc = ac0 + ac1;
    int col = nt * 16 + colL;
    float bias = bg[col];
#pragma unroll
    for (int r = 0; r < 8; ++r) {
      int row = m0 + hi * 8 + r;
      float g = acc[r] + bias;
      gSig[row * 264 + col] = 1.0f / (1.0f + __expf(-g));
    }
  }
  __syncthreads();

  // ---- Phase 5: vr = (vh @ Wu^T) * sigmoid(gate) -> global [N, V_OUT, 3]
  //      Wu B-tiles shared by the 3 components: preload once, reuse 3x.
#pragma unroll 1
  for (int nt8 = 0; nt8 < 8; ++nt8) {
    int nt = ch * 8 + nt8;
    v16bf bq[8];
#pragma unroll
    for (int kt = 0; kt < 8; ++kt) bq[kt] = glb_loadB(wsWu, nt * 8 + kt, lane);
    int col = nt * 16 + colL;
#pragma unroll 1
    for (int c = 0; c < 3; ++c) {
      const __bf16* A = sVH + c * 16896;
      v8f ac0 = V8F_ZERO, ac1 = V8F_ZERO;
#pragma unroll
      for (int kt = 0; kt < 8; kt += 2) {
        v16bf a0 = lds_loadA(A, 264, m0, kt * 32, lane);
        v16bf a1 = lds_loadA(A, 264, m0, (kt + 1) * 32, lane);
        ac0 = WMMA_BF16(a0, bq[kt], ac0);
        ac1 = WMMA_BF16(a1, bq[kt + 1], ac1);
      }
      v8f acc = ac0 + ac1;
#pragma unroll
      for (int r = 0; r < 8; ++r) {
        int row = m0 + hi * 8 + r;
        if (n0 + row < N) {
          float val = acc[r] * gSig[row * 264 + col];
          out_vr[((long)(n0 + row) * 256 + col) * 3 + c] = val;
        }
      }
    }
  }
}

// ---------------------------------------------------------------------------
// Launch: convert/swizzle weights into d_ws (bf16, 1.25 MB), then fused kernel.
// ---------------------------------------------------------------------------
extern "C" void kernel_launch(void* const* d_in, const int* in_sizes, int n_in,
                              void* d_out, int out_size, void* d_ws, size_t ws_size,
                              hipStream_t stream) {
  const float* s  = (const float*)d_in[0];
  const float* v  = (const float*)d_in[1];
  const float* Wd = (const float*)d_in[2];
  const float* Ws = (const float*)d_in[3];
  const float* bs = (const float*)d_in[4];
  const float* Wu = (const float*)d_in[5];
  const float* Wg = (const float*)d_in[6];
  const float* bg = (const float*)d_in[7];
  const int N = in_sizes[0] / 512;

  __bf16* wsBase = (__bf16*)d_ws;
  __bf16* wsWd = wsBase;           // 256x256 -> 65536
  __bf16* wsWs = wsBase + 65536;   // 512x768 -> 393216
  __bf16* wsWu = wsBase + 458752;  // 256x256 -> 65536
  __bf16* wsWg = wsBase + 524288;  // 256x512 -> 131072

  cvt_weights<<<65536 / 256, 256, 0, stream>>>(Wd, wsWd, 256, 8, 65536);
  cvt_weights<<<393216 / 256, 256, 0, stream>>>(Ws, wsWs, 768, 24, 393216);
  cvt_weights<<<65536 / 256, 256, 0, stream>>>(Wu, wsWu, 256, 8, 65536);
  cvt_weights<<<131072 / 256, 256, 0, stream>>>(Wg, wsWg, 512, 16, 131072);

  float* out_sr = (float*)d_out;
  float* out_vr = out_sr + (long)N * 512;

  const int kLds = 303104;
  (void)hipFuncSetAttribute((const void*)gvp_fused,
                            hipFuncAttributeMaxDynamicSharedMemorySize, kLds);
  int blocks = (N + 63) / 64;
  gvp_fused<<<blocks, 256, kLds, stream>>>(s, v, bs, bg, wsWd, wsWs, wsWu, wsWg,
                                           out_sr, out_vr, N);
}
